// Transformer_53927609368731
// MI455X (gfx1250) — matrix operand order repaired and checked by reading, not compiled, not run on hardware
//
#include <hip/hip_runtime.h>

typedef unsigned short u16;
typedef __attribute__((ext_vector_type(16))) __bf16 v16bf;
typedef __attribute__((ext_vector_type(8)))  float  v8f;
typedef __attribute__((ext_vector_type(4)))  unsigned int v4u;
typedef __attribute__((ext_vector_type(8)))  int v8i;
typedef __attribute__((ext_vector_type(4)))  int v4i;

#define BSZ 2
#define QS 2048
#define HID 1024
#define NH 16
#define HD 64

__device__ __forceinline__ u16 f32_to_bf16(float f) {
  union { float f; unsigned int u; } x; x.f = f;
  unsigned int r = x.u + 0x7FFFu + ((x.u >> 16) & 1u);
  return (u16)(r >> 16);
}

__device__ __forceinline__ v8f v8f_zero() {
  v8f z;
#pragma unroll
  for (int e = 0; e < 8; ++e) z[e] = 0.f;
  return z;
}

__device__ __forceinline__ v8f wmma_bf16(v16bf a, v16bf b, v8f c) {
  return __builtin_amdgcn_wmma_f32_16x16x32_bf16(false, a, false, b, (short)0, c,
                                                 false, false);
}

// Low 32 bits of a generic LDS pointer are the DS-space offset (aperture form).
__device__ __forceinline__ unsigned lds_off(const void* p) {
  return (unsigned)(size_t)p;
}

// ---- Tensor Data Mover: 2D tile (bf16 elements) global -> LDS ----
// tile0 = contiguous elems per row, tile1 = rows, stride0 = row pitch (elems).
// D# per CDNA5 ISA ch8: group0 {count=1,lds,global,type=2}, group1 {dsize=2B,
// tensor dims (huge: tiles are in-bounds), tile dims, dim0 stride}. groups2/3=0.
__device__ __forceinline__ void tdm_load_tile(unsigned ldsaddr, const void* gp,
                                              unsigned tile0, unsigned tile1,
                                              unsigned long long stride0) {
  unsigned long long ga = (unsigned long long)(size_t)gp;
  v4u g0; v8i g1; v4i g2; v4i g3;
  g0[0] = 1u;                                         // count=1, user D#
  g0[1] = ldsaddr;                                    // lds_addr (bytes)
  g0[2] = (unsigned)ga;                               // global_addr[31:0]
  g0[3] = (unsigned)((ga >> 32) & 0x1FFFFFFu) | (2u << 30);  // [56:32] + type=2
  g1[0] = (int)(1u << 16);                            // data_size=1 (2 bytes)
  g1[1] = (int)0xFFFF0000u;                           // tensor_dim0 lo16 (0x7FFFFFFF)
  g1[2] = (int)0xFFFF7FFFu;                           // dim0 hi16 | dim1 lo16
  g1[3] = (int)(0x7FFFu | (tile0 << 16));             // dim1 hi16 | tile_dim0
  g1[4] = (int)(tile1 & 0xFFFFu);                     // tile_dim1 (tile_dim2=0)
  g1[5] = (int)(unsigned)(stride0 & 0xFFFFFFFFull);   // tensor_dim0_stride lo32
  g1[6] = (int)(unsigned)((stride0 >> 32) & 0xFFFFull);
  g1[7] = 0;
#pragma unroll
  for (int i = 0; i < 4; ++i) { g2[i] = 0; g3[i] = 0; }
#if defined(__clang_major__) && (__clang_major__ >= 23)
  v8i g4;
#pragma unroll
  for (int i = 0; i < 8; ++i) g4[i] = 0;
  __builtin_amdgcn_tensor_load_to_lds(g0, g1, g2, g3, g4, 0);
#else
  __builtin_amdgcn_tensor_load_to_lds(g0, g1, g2, g3, 0);
#endif
}

union frag_u { uint4 u[2]; v16bf v; };

// A fragment (16x32 bf16 MxK), base row-major [M][stride].
__device__ __forceinline__ v16bf load_a_frag(const u16* base, int stride, int m0,
                                             int k0, int lane) {
  const u16* p = base + (m0 + (lane & 15)) * stride + k0 + ((lane < 16) ? 0 : 8);
  frag_u t;
  t.u[0] = *(const uint4*)(p);
  t.u[1] = *(const uint4*)(p + 16);
  return t.v;
}

// B fragment (32x16 bf16 KxN) from B^T row-major [N][stride].
__device__ __forceinline__ v16bf load_b_frag(const u16* base, int stride, int n0,
                                             int k0, int lane) {
  const u16* p = base + (n0 + (lane & 15)) * stride + k0 + ((lane < 16) ? 0 : 8);
  frag_u t;
  t.u[0] = *(const uint4*)(p);
  t.u[1] = *(const uint4*)(p + 16);
  return t.v;
}

// ---------------- converts ----------------
__global__ __launch_bounds__(256) void cvt_kernel(const float* __restrict__ src,
                                                  u16* __restrict__ dst, int n) {
  for (int i = blockIdx.x * 256 + threadIdx.x; i < n; i += gridDim.x * 256)
    dst[i] = f32_to_bf16(src[i]);
}

__global__ __launch_bounds__(256) void cvtT_kernel(const float* __restrict__ src,
                                                   u16* __restrict__ dst, int K,
                                                   int N) {
  for (int i = blockIdx.x * 256 + threadIdx.x; i < K * N; i += gridDim.x * 256) {
    int r = i / N, c = i - r * N;
    dst[c * K + r] = f32_to_bf16(src[i]);
  }
}

// ---------------- WMMA GEMM, TDM double-buffered: C = A[M,K] * Bt[N,K]^T ----
// MODE 0: q -> p0=bf16(v+f0[c]) / p1=bf16(v+f1[c]) per-head [BH][QS][HD]
// MODE 1: k -> per-head [BH][QS][HD];  MODE 2: v -> per-head [BH][HD][QS]
// MODE 3: r -> reversed per-head [NH][QS][HD];  MODE 4: f32 [M][N]
template <int MODE>
__global__ __launch_bounds__(128) void gemm_bf16(
    const u16* __restrict__ A, const u16* __restrict__ Bt, int M, int N, int K,
    u16* __restrict__ p0, u16* __restrict__ p1, float* __restrict__ fp,
    const float* __restrict__ f0, const float* __restrict__ f1) {
  __shared__ u16 As[2][64 * 32];
  __shared__ u16 Bs[2][64 * 32];
  const int tid = threadIdx.x;
  const int wave = tid >> 5, lane = tid & 31;
  const int bm = blockIdx.x * 64, bn = blockIdx.y * 64;
  const u16* Abase = A + (size_t)bm * K;
  const u16* Bbase = Bt + (size_t)bn * K;
  v8f acc[4];
#pragma unroll
  for (int i = 0; i < 4; ++i) acc[i] = v8f_zero();

  if (wave == 0) {
    tdm_load_tile(lds_off(&As[0][0]), Abase, 32, 64, (unsigned long long)K);
    tdm_load_tile(lds_off(&Bs[0][0]), Bbase, 32, 64, (unsigned long long)K);
    __builtin_amdgcn_s_wait_tensorcnt(0);
  }
  __syncthreads();

  int cur = 0;
  for (int k0 = 0; k0 < K; k0 += 32) {
    const int nxt = cur ^ 1;
    const bool more = (k0 + 32) < K;
    if (more && wave == 0) {  // prefetch next chunk while computing this one
      tdm_load_tile(lds_off(&As[nxt][0]), Abase + k0 + 32, 32, 64,
                    (unsigned long long)K);
      tdm_load_tile(lds_off(&Bs[nxt][0]), Bbase + k0 + 32, 32, 64,
                    (unsigned long long)K);
    }
    v16bf a = load_a_frag(As[cur], 32, wave * 16, 0, lane);
#pragma unroll
    for (int ns = 0; ns < 4; ++ns) {
      v16bf b = load_b_frag(Bs[cur], 32, ns * 16, 0, lane);
      acc[ns] = wmma_bf16(a, b, acc[ns]);
    }
    if (more && wave == 0) __builtin_amdgcn_s_wait_tensorcnt(0);
    __syncthreads();
    cur = nxt;
  }

  const int lc = lane & 15;
  const int rb = (lane < 16) ? 0 : 8;
#pragma unroll
  for (int ns = 0; ns < 4; ++ns) {
    const int c = bn + ns * 16 + lc;
#pragma unroll
    for (int e = 0; e < 8; ++e) {
      const int m = bm + wave * 16 + rb + e;
      const float v = acc[ns][e];
      if constexpr (MODE == 0) {
        const int b = m >> 11, s = m & (QS - 1), n = c >> 6, d = c & (HD - 1);
        const size_t idx = (((size_t)(b * NH + n) * QS) + s) * HD + d;
        p0[idx] = f32_to_bf16(v + f0[c]);
        p1[idx] = f32_to_bf16(v + f1[c]);
      } else if constexpr (MODE == 1) {
        const int b = m >> 11, s = m & (QS - 1), n = c >> 6, d = c & (HD - 1);
        p0[(((size_t)(b * NH + n) * QS) + s) * HD + d] = f32_to_bf16(v);
      } else if constexpr (MODE == 2) {
        const int b = m >> 11, s = m & (QS - 1), n = c >> 6, d = c & (HD - 1);
        p0[(((size_t)(b * NH + n) * HD) + d) * QS + s] = f32_to_bf16(v);
      } else if constexpr (MODE == 3) {
        const int n = c >> 6, d = c & (HD - 1);
        p0[((size_t)n * QS + (QS - 1 - m)) * HD + d] = f32_to_bf16(v);
      } else {
        fp[(size_t)m * HID + c] = v;
      }
    }
  }
}

// ---------------- fused flash attention with rel-shift band ----------------
// S[q,k] = scale*( qrw[q].k[k] + qrr[q].rrev[q-k] ), k<=q; online softmax; O=P*V
__global__ __launch_bounds__(128) void attn_kernel(
    const u16* __restrict__ qrw, const u16* __restrict__ qrr,
    const u16* __restrict__ kh, const u16* __restrict__ vt,
    const u16* __restrict__ rrev, u16* __restrict__ att) {
  extern __shared__ char smem[];
  u16* Qw = (u16*)smem;                 // 64x64 bf16
  u16* Qr = Qw + 64 * 64;               // 64x64
  u16* Ks = Qr + 64 * 64;               // 64x64 (K rows x d)
  u16* Vs = Ks + 64 * 64;               // 64x64 (V^T: d rows x k)
  u16* Ps = Vs + 64 * 64;               // 64x64 probs bf16
  u16* Rs = Ps + 64 * 64;               // 128x64 (rrev band rows x d)
  float* Sb = (float*)(Rs + 128 * 64);  // 64x64 scores
  float* C2 = Sb + 64 * 64;             // 64x128 band GEMM result
  float* rowm = C2 + 64 * 128;          // 64
  float* rowl = rowm + 64;              // 64
  float* rowa = rowl + 64;              // 64 (per-row rescale alpha)

  const int tid = threadIdx.x;
  const int wave = tid >> 5, lane = tid & 31;
  const int m0 = wave * 16;
  const int lc = lane & 15;
  const int rbase = m0 + ((lane < 16) ? 0 : 8);
  const int bh = blockIdx.x;            // b*NH + n
  const int q0 = blockIdx.y * 64;
  const int n = bh & (NH - 1);
  const size_t headq = (size_t)bh * QS * HD;
  const size_t headv = (size_t)bh * HD * QS;
  const size_t headr = (size_t)n * QS * HD;
  const float NEGINF = -__builtin_inff();

  if (wave == 0) {  // stage Q tiles (contiguous 64x64) via TDM
    tdm_load_tile(lds_off(Qw), qrw + headq + (size_t)q0 * HD, 4096, 1, 4096);
    tdm_load_tile(lds_off(Qr), qrr + headq + (size_t)q0 * HD, 4096, 1, 4096);
    __builtin_amdgcn_s_wait_tensorcnt(0);
  }
  if (tid < 64) { rowm[tid] = NEGINF; rowl[tid] = 0.f; }
  __syncthreads();

  // Q fragments are k-loop invariant: hoist.
  v16bf aw0 = load_a_frag(Qw, 64, m0, 0, lane);
  v16bf aw1 = load_a_frag(Qw, 64, m0, 32, lane);
  v16bf ar0 = load_a_frag(Qr, 64, m0, 0, lane);
  v16bf ar1 = load_a_frag(Qr, 64, m0, 32, lane);

  v8f oacc[4];  // O accumulator in registers (WMMA C layout)
#pragma unroll
  for (int i = 0; i < 4; ++i) oacc[i] = v8f_zero();

  const int nkb = (q0 >> 6) + 1;
  for (int kb = 0; kb < nkb; ++kb) {
    const int k0 = kb * 64;
    const int dbase = q0 - k0;
    if (wave == 0) {
      tdm_load_tile(lds_off(Ks), kh + headq + (size_t)k0 * HD, 4096, 1, 4096);
      tdm_load_tile(lds_off(Vs), vt + headv + k0, 64, 64, QS);
      if (dbase >= 63)  // off-diagonal: band rows contiguous, no clamp needed
        tdm_load_tile(lds_off(Rs), rrev + headr + (size_t)(dbase - 63) * HD,
                      8192, 1, 8192);
    }
    if (dbase < 63) {  // diagonal block: clamped band rows, manual stage
      for (int i = tid; i < 1024; i += 128) {
        int t = i >> 3, c8 = (i & 7) * 8;
        int idx = dbase - 63 + t;
        idx = idx < 0 ? 0 : idx;  // clamped rows correspond to masked entries
        *(uint4*)&Rs[t * 64 + c8] =
            *(const uint4*)&rrev[headr + (size_t)idx * HD + c8];
      }
    }
    if (wave == 0) __builtin_amdgcn_s_wait_tensorcnt(0);
    __syncthreads();

    // AC tile: Qw @ Ks^T -> Sb[64x64]
#pragma unroll
    for (int ns = 0; ns < 4; ++ns) {
      v8f acc = v8f_zero();
      acc = wmma_bf16(aw0, load_b_frag(Ks, 64, ns * 16, 0, lane), acc);
      acc = wmma_bf16(aw1, load_b_frag(Ks, 64, ns * 16, 32, lane), acc);
      const int col = ns * 16 + lc;
#pragma unroll
      for (int e = 0; e < 8; ++e) Sb[(rbase + e) * 64 + col] = acc[e];
    }
    // BD band: Qr @ Rs^T -> C2[64x128]; BD[i][j] = C2[i][i-j+63]
#pragma unroll
    for (int ns = 0; ns < 8; ++ns) {
      v8f acc = v8f_zero();
      acc = wmma_bf16(ar0, load_b_frag(Rs, 64, ns * 16, 0, lane), acc);
      acc = wmma_bf16(ar1, load_b_frag(Rs, 64, ns * 16, 32, lane), acc);
      const int col = ns * 16 + lc;
#pragma unroll
      for (int e = 0; e < 8; ++e) C2[(rbase + e) * 128 + col] = acc[e];
    }
    __syncthreads();

    // online softmax, one thread per row
    if (tid < 64) {
      const int i = tid, q = q0 + i;
      float mloc = NEGINF;
      for (int j = 0; j < 64; ++j) {
        float s;
        if (k0 + j > q) s = NEGINF;
        else s = (Sb[i * 64 + j] + C2[i * 128 + (i - j + 63)]) * 0.125f;
        Sb[i * 64 + j] = s;
        mloc = fmaxf(mloc, s);
      }
      const float mold = rowm[i];
      const float mnew = fmaxf(mold, mloc);
      const float alpha = __expf(mold - mnew);  // exp(-inf)=0 on first block
      float sum = 0.f;
      for (int j = 0; j < 64; ++j) {
        const float p = __expf(Sb[i * 64 + j] - mnew);
        Ps[i * 64 + j] = f32_to_bf16(p);
        sum += p;
      }
      rowa[i] = alpha;
      rowl[i] = rowl[i] * alpha + sum;
      rowm[i] = mnew;
    }
    __syncthreads();

    // O = O*alpha + P @ V (register accumulators; Vs = V^T is B^T directly)
    {
      float al[8];
#pragma unroll
      for (int e = 0; e < 8; ++e) al[e] = rowa[rbase + e];
      v16bf p0f = load_a_frag(Ps, 64, m0, 0, lane);
      v16bf p1f = load_a_frag(Ps, 64, m0, 32, lane);
#pragma unroll
      for (int ns = 0; ns < 4; ++ns) {
        v8f pv = v8f_zero();
        pv = wmma_bf16(p0f, load_b_frag(Vs, 64, ns * 16, 0, lane), pv);
        pv = wmma_bf16(p1f, load_b_frag(Vs, 64, ns * 16, 32, lane), pv);
#pragma unroll
        for (int e = 0; e < 8; ++e) oacc[ns][e] = oacc[ns][e] * al[e] + pv[e];
      }
    }
    __syncthreads();
  }

  // epilogue: divide by softmax denominator, store bf16 directly to global
  const int b = bh >> 4;
  float li[8];
#pragma unroll
  for (int e = 0; e < 8; ++e) li[e] = rowl[rbase + e];
#pragma unroll
  for (int ns = 0; ns < 4; ++ns) {
    const int col = n * HD + ns * 16 + lc;
#pragma unroll
    for (int e = 0; e < 8; ++e)
      att[(size_t)(b * QS + q0 + rbase + e) * (NH * HD) + col] =
          f32_to_bf16(oacc[ns][e] / li[e]);
  }
}

// ---------------- residual + LayerNorm ----------------
__global__ __launch_bounds__(256) void ln_kernel(const float* __restrict__ x,
                                                 const float* __restrict__ pre,
                                                 const float* __restrict__ gamma,
                                                 const float* __restrict__ beta,
                                                 float* __restrict__ out) {
  __shared__ float red[256];
  const int row = blockIdx.x, tid = threadIdx.x;
  const float* xr = x + (size_t)row * HID;
  const float* pr = pre + (size_t)row * HID;
  float s = 0.f;
  for (int c = tid; c < HID; c += 256) s += xr[c] + pr[c];
  red[tid] = s;
  __syncthreads();
  for (int off = 128; off > 0; off >>= 1) {
    if (tid < off) red[tid] += red[tid + off];
    __syncthreads();
  }
  const float mu = red[0] / (float)HID;
  __syncthreads();
  float vs = 0.f;
  for (int c = tid; c < HID; c += 256) {
    const float v = xr[c] + pr[c] - mu;
    vs += v * v;
  }
  red[tid] = vs;
  __syncthreads();
  for (int off = 128; off > 0; off >>= 1) {
    if (tid < off) red[tid] += red[tid + off];
    __syncthreads();
  }
  const float rstd = rsqrtf(red[0] / (float)HID + 1e-5f);
  for (int c = tid; c < HID; c += 256)
    out[(size_t)row * HID + c] = (xr[c] + pr[c] - mu) * rstd * gamma[c] + beta[c];
}

extern "C" void kernel_launch(void* const* d_in, const int* in_sizes, int n_in,
                              void* d_out, int out_size, void* d_ws, size_t ws_size,
                              hipStream_t stream) {
  (void)in_sizes; (void)n_in; (void)out_size; (void)ws_size;
  const float* x   = (const float*)d_in[0];
  // d_in[1] = mask (applied analytically)
  const float* pe  = (const float*)d_in[2];
  const float* rrb = (const float*)d_in[3];
  const float* rwb = (const float*)d_in[4];
  const float* Wq  = (const float*)d_in[5];
  const float* Wk  = (const float*)d_in[6];
  const float* Wv  = (const float*)d_in[7];
  const float* Wr  = (const float*)d_in[8];
  const float* Wo  = (const float*)d_in[9];
  const float* gam = (const float*)d_in[10];
  const float* bet = (const float*)d_in[11];
  float* out = (float*)d_out;

  char* W = (char*)d_ws;
  size_t off = 0;
  auto alloc = [&](size_t bytes) {
    char* p = W + off;
    off = (off + bytes + 255) & ~(size_t)255;
    return p;
  };
  const size_t NX = (size_t)BSZ * QS * HID;
  const size_t NP = (size_t)QS * HID;
  const size_t NW = (size_t)HID * HID;
  u16* xbf   = (u16*)alloc(NX * 2);
  u16* pebf  = (u16*)alloc(NP * 2);
  u16* WqT   = (u16*)alloc(NW * 2);
  u16* WkT   = (u16*)alloc(NW * 2);
  u16* WvT   = (u16*)alloc(NW * 2);
  u16* WrT   = (u16*)alloc(NW * 2);
  u16* WoT   = (u16*)alloc(NW * 2);
  u16* qrw   = (u16*)alloc(NX * 2);
  u16* qrr   = (u16*)alloc(NX * 2);
  u16* khead = (u16*)alloc(NX * 2);
  u16* vth   = (u16*)alloc(NX * 2);
  u16* rrevb = (u16*)alloc(NP * 2);
  u16* attb  = (u16*)alloc(NX * 2);
  float* pre = (float*)alloc(NX * 4);

  cvt_kernel<<<2048, 256, 0, stream>>>(x, xbf, (int)NX);
  cvt_kernel<<<1024, 256, 0, stream>>>(pe, pebf, (int)NP);
  cvtT_kernel<<<1024, 256, 0, stream>>>(Wq, WqT, HID, HID);
  cvtT_kernel<<<1024, 256, 0, stream>>>(Wk, WkT, HID, HID);
  cvtT_kernel<<<1024, 256, 0, stream>>>(Wv, WvT, HID, HID);
  cvtT_kernel<<<1024, 256, 0, stream>>>(Wr, WrT, HID, HID);
  cvtT_kernel<<<1024, 256, 0, stream>>>(Wo, WoT, HID, HID);

  const dim3 blk(128);
  gemm_bf16<0><<<dim3(64, 16), blk, 0, stream>>>(xbf, WqT, 4096, 1024, 1024,
                                                 qrw, qrr, nullptr, rwb, rrb);
  gemm_bf16<1><<<dim3(64, 16), blk, 0, stream>>>(xbf, WkT, 4096, 1024, 1024,
                                                 khead, nullptr, nullptr, nullptr, nullptr);
  gemm_bf16<2><<<dim3(64, 16), blk, 0, stream>>>(xbf, WvT, 4096, 1024, 1024,
                                                 vth, nullptr, nullptr, nullptr, nullptr);
  gemm_bf16<3><<<dim3(32, 16), blk, 0, stream>>>(pebf, WrT, 2048, 1024, 1024,
                                                 rrevb, nullptr, nullptr, nullptr, nullptr);

  const size_t smem = (size_t)(64 * 64 * 5 + 128 * 64) * 2 +
                      (size_t)(64 * 64 + 64 * 128 + 3 * 64) * 4;  // 107264 B
  attn_kernel<<<dim3(BSZ * NH, QS / 64), blk, smem, stream>>>(qrw, qrr, khead,
                                                              vth, rrevb, attb);

  gemm_bf16<4><<<dim3(64, 16), blk, 0, stream>>>(attb, WoT, 4096, 1024, 1024,
                                                 nullptr, nullptr, pre, nullptr, nullptr);
  ln_kernel<<<BSZ * QS, 256, 0, stream>>>(x, pre, gam, bet, out);
}
